// RelativeMultiHeadAttn_48361331753684
// MI455X (gfx1250) — matrix operand order repaired and checked by reading, not compiled, not run on hardware
//
#include <hip/hip_runtime.h>
#include <math.h>

// ---------------------------------------------------------------------------
// CDNA5 (gfx1250) wave32 WMMA types / helpers
// ---------------------------------------------------------------------------
typedef __attribute__((ext_vector_type(16))) __bf16 v16bf;
typedef __attribute__((ext_vector_type(8)))  float  v8f;
typedef __attribute__((ext_vector_type(4)))  int    v4i;

#define Bq 4
#define Lq 1024
#define Dm 1024
#define Hh 16
#define Dk 64
#define Pp 2048
#define HR_ROWS 2064   // 2048 + 16 pad rows so the 48-row pos slab never reads OOB

#define WMMA_BF16(a, b, c) \
  __builtin_amdgcn_wmma_f32_16x16x32_bf16(false, (a), false, (b), (short)0, (c), false, false)

// A fragment: 16(M) x 32(K) tile, row-major source, layout:
//   lanes 0-15  : M=lane,    K 0-15  (contiguous)
//   lanes 16-31 : M=lane-16, K 16-31 (contiguous)
static __device__ __forceinline__ v16bf load_a_f32(const float* src, int ld, int lane) {
  const int m = lane & 15, sel = (lane >> 4) & 1;
  const float* p = src + (size_t)m * ld + sel * 16;
  v16bf a;
#pragma unroll
  for (int i = 0; i < 16; ++i) a[i] = (__bf16)p[i];
  return a;
}
static __device__ __forceinline__ v16bf load_a_bf16(const __bf16* src, int ld, int lane) {
  const int m = lane & 15, sel = (lane >> 4) & 1;
  const __bf16* p = src + (size_t)m * ld + sel * 16;
  v16bf a;
#pragma unroll
  for (int i = 0; i < 16; ++i) a[i] = p[i];
  return a;
}
// B fragment: 32(K) x 16(N) tile, loaded from B^T stored row-major [N][K]:
//   lanes 0-15  : N=lane,    K 0-15  (contiguous)
//   lanes 16-31 : N=lane-16, K 16-31 (contiguous)
static __device__ __forceinline__ v16bf load_b_f32(const float* srcT, int ld, int lane) {
  const int n = lane & 15, sel = (lane >> 4) & 1;
  const float* p = srcT + (size_t)n * ld + sel * 16;
  v16bf b;
#pragma unroll
  for (int i = 0; i < 16; ++i) b[i] = (__bf16)p[i];
  return b;
}
static __device__ __forceinline__ v16bf load_b_bf16(const __bf16* srcT, int ld, int lane) {
  const int n = lane & 15, sel = (lane >> 4) & 1;
  const __bf16* p = srcT + (size_t)n * ld + sel * 16;
  v16bf b;
#pragma unroll
  for (int i = 0; i < 16; ++i) b[i] = p[i];
  return b;
}
// C/D layout: VGPR r, lane l -> M = r + 8*(l>>4), N = l&15

// ---------------------------------------------------------------------------
// gfx1250 async global->LDS copy (ASYNCcnt path): one instruction moves
// 32 lanes x 16B. Builtin signature (from toolchain diagnostic): first param
// is non-const v4i* in device/global AS1; LDS side is v4i* in AS3.
// ---------------------------------------------------------------------------
static __device__ __forceinline__ void async_cp16(unsigned lds_off, const char* g) {
#if __has_builtin(__builtin_amdgcn_global_load_async_to_lds_b128)
  typedef __attribute__((address_space(1))) v4i* as1_v4i;
  typedef __attribute__((address_space(3))) v4i* as3_v4i;
  __builtin_amdgcn_global_load_async_to_lds_b128(
      (as1_v4i)(uintptr_t)g, (as3_v4i)(uintptr_t)lds_off, 0, 0);
#else
  asm volatile("global_load_async_to_lds_b128 %0, %1, off"
               :: "v"(lds_off), "v"(g) : "memory");
#endif
}
static __device__ __forceinline__ void wait_async_le4() {
#if __has_builtin(__builtin_amdgcn_s_wait_asynccnt)
  __builtin_amdgcn_s_wait_asynccnt(4);
#else
  asm volatile("s_wait_asynccnt 0x4" ::: "memory");
#endif
}
static __device__ __forceinline__ void wait_async_0() {
#if __has_builtin(__builtin_amdgcn_s_wait_asynccnt)
  __builtin_amdgcn_s_wait_asynccnt(0);
#else
  asm volatile("s_wait_asynccnt 0x0" ::: "memory");
#endif
}

// ---------------------------------------------------------------------------
// Sinusoidal relative position table r[P, Dm], interleaved sin/cos, *INIT*DW
// ---------------------------------------------------------------------------
__global__ void relpos_kernel(float* __restrict__ r) {
  const int p = blockIdx.x;
  const float pos = (float)(Lq - p);            // arange(Lq, -Lk, -1)
  for (int f = threadIdx.x; f < Dm / 2; f += blockDim.x) {
    const float freq = powf(10000.0f, -(2.0f * (float)f) / (float)Dm);
    const float ang = pos * freq;
    r[(size_t)p * Dm + 2 * f]     = sinf(ang) * 102.4f;   // INIT*DW = 0.1*1024
    r[(size_t)p * Dm + 2 * f + 1] = cosf(ang) * 102.4f;
  }
}

// ---------------------------------------------------------------------------
// Per-head projections. mode 0: hq -> (hq+u_w, hq+v_w); 1: hk; 2: hv (transposed)
// One wave = one 16x16 tile, 4 waves/block cover N=64, K-loop over Dm.
// ---------------------------------------------------------------------------
__global__ __launch_bounds__(128) void proj_qkv_kernel(
    const float* __restrict__ X, const float* __restrict__ W,
    const float* __restrict__ u_w, const float* __restrict__ v_w,
    __bf16* __restrict__ out0, __bf16* __restrict__ out1, int mode) {
  const int lane = threadIdx.x & 31, wave = threadIdx.x >> 5;
  int t = blockIdx.x;
  const int mt = t % (Lq / 16); t /= (Lq / 16);
  const int h = t % Hh, b = t / Hh;
  const int bh = b * Hh + h;
  const int n0 = wave * 16;
  const float* A  = X + (size_t)(b * Lq + mt * 16) * Dm;
  const float* BT = W + (size_t)(h * Dk + n0) * Dm;   // W[h] is [out=N][in=K] == B^T
  v8f c = {};
  for (int k0 = 0; k0 < Dm; k0 += 32)
    c = WMMA_BF16(load_a_f32(A + k0, Dm, lane), load_b_f32(BT + k0, Dm, lane), c);
  const int half = (lane >> 4) & 1, n = lane & 15;
#pragma unroll
  for (int r = 0; r < 8; ++r) {
    const int row = mt * 16 + r + 8 * half, col = n0 + n;
    if (mode == 0) {
      out0[((size_t)bh * Lq + row) * Dk + col] = (__bf16)(c[r] + u_w[col]);
      out1[((size_t)bh * Lq + row) * Dk + col] = (__bf16)(c[r] + v_w[col]);
    } else if (mode == 1) {
      out0[((size_t)bh * Lq + row) * Dk + col] = (__bf16)c[r];
    } else { // hv^T : [bh][dv][l] so PV-GEMM B-operand is contiguous in K=j
      out0[((size_t)bh * Dk + col) * Lq + row] = (__bf16)c[r];
    }
  }
}

__global__ __launch_bounds__(128) void proj_r_kernel(
    const float* __restrict__ R, const float* __restrict__ Wr, __bf16* __restrict__ hr) {
  const int lane = threadIdx.x & 31, wave = threadIdx.x >> 5;
  const int mt = blockIdx.x % (Pp / 16), h = blockIdx.x / (Pp / 16);
  const int n0 = wave * 16;
  const float* A  = R + (size_t)(mt * 16) * Dm;
  const float* BT = Wr + (size_t)(h * Dk + n0) * Dm;
  v8f c = {};
  for (int k0 = 0; k0 < Dm; k0 += 32)
    c = WMMA_BF16(load_a_f32(A + k0, Dm, lane), load_b_f32(BT + k0, Dm, lane), c);
  const int half = (lane >> 4) & 1, n = lane & 15;
#pragma unroll
  for (int r = 0; r < 8; ++r)
    hr[((size_t)h * HR_ROWS + mt * 16 + r + 8 * half) * Dk + n0 + n] = (__bf16)c[r];
}

// ---------------------------------------------------------------------------
// Fused flash attention with TXL relative-position term.
// Block = 4 waves sharing (b,h); each wave owns one 16-row query tile and
// streams 32 keys/iter. K/V tiles are staged into LDS once per block with
// double-buffered async global->LDS copies (ASYNCcnt + barriers).
// score[i,j] = ((hq[i]+u)·hk[j] + (hq[i]+v)·hr[L+j-i]) / 32
// Pos term: unshifted 16x48 panel via WMMA, Toeplitz gather through LDS.
// ---------------------------------------------------------------------------
__global__ __launch_bounds__(128) void flash_kernel(
    const __bf16* __restrict__ hq_u, const __bf16* __restrict__ hq_v,
    const __bf16* __restrict__ hk,   const __bf16* __restrict__ hvT,
    const __bf16* __restrict__ hr,   __bf16* __restrict__ heads) {
  __shared__ __bf16 ldsK[2][32][64];     // keys j0..j0+31, [j][dk]      (4KB x2)
  __shared__ __bf16 ldsV[2][64][32];     // values [dv][j-j0]            (4KB x2)
  __shared__ float  ldsP[4][16][48];     // per-wave unshifted pos panel
  __shared__ __bf16 ldsProb[4][16][32];  // per-wave probs (C->A relayout)
  const int tid = threadIdx.x;
  const int lane = tid & 31, wave = tid >> 5;
  const int qblk = blockIdx.x % (Lq / 64);
  const int bh   = blockIdx.x / (Lq / 64);
  const int b = bh / Hh, h = bh % Hh;
  const int i0 = qblk * 64 + wave * 16;

  const __bf16* Aqu = hq_u + ((size_t)bh * Lq + i0) * Dk;
  const __bf16* Aqv = hq_v + ((size_t)bh * Lq + i0) * Dk;
  const __bf16* Kb  = hk  + (size_t)bh * Lq * Dk;
  const __bf16* VT  = hvT + (size_t)bh * Dk * Lq;
  const __bf16* Hr  = hr  + (size_t)h * HR_ROWS * Dk;

  const unsigned ldsK_off = (unsigned)(uintptr_t)&ldsK[0][0][0];
  const unsigned ldsV_off = (unsigned)(uintptr_t)&ldsV[0][0][0];

  // Stage one 32-key block (K: 4KB contiguous, V: 64 rows of 64B) -> LDS[buf].
  auto stage_kv = [&](int buf, int j0) {
    const char* kg = (const char*)(Kb + (size_t)j0 * Dk);
#pragma unroll
    for (int i = 0; i < 2; ++i) {
      const int c = tid + 128 * i;                       // 16B chunk id, 0..255
      async_cp16(ldsK_off + buf * 4096 + c * 16, kg + c * 16);
    }
    const char* vg = (const char*)(VT + j0);
#pragma unroll
    for (int i = 0; i < 2; ++i) {
      const int c = tid + 128 * i;
      const int dv = c >> 2, part = c & 3;               // 4 chunks per 64B row
      async_cp16(ldsV_off + buf * 4096 + dv * 64 + part * 16,
                 vg + (size_t)dv * (Lq * 2) + part * 16);
    }
  };

  const v16bf aqu0 = load_a_bf16(Aqu, Dk, lane),  aqu1 = load_a_bf16(Aqu + 32, Dk, lane);
  const v16bf aqv0 = load_a_bf16(Aqv, Dk, lane),  aqv1 = load_a_bf16(Aqv + 32, Dk, lane);

  v8f O0 = {}, O1 = {}, O2 = {}, O3 = {};
  float mi[8], li[8];
#pragma unroll
  for (int r = 0; r < 8; ++r) { mi[r] = -INFINITY; li[r] = 0.f; }
  const int half = (lane >> 4) & 1, n = lane & 15;
  const float rcpT = 0.03125f;  // 1/sqrt(1024)

  stage_kv(0, 0);  // prologue fill of buffer 0

  for (int it = 0; it < Lq / 32; ++it) {
    const int j0 = it * 32;
    const int cur = it & 1;
    const bool more = (it + 1) < (Lq / 32);
    if (more) {
      stage_kv(cur ^ 1, j0 + 32);   // overlap next copy with current compute
      wait_async_le4();             // just-issued 4 stay in flight; buf[cur] done
    } else {
      wait_async_0();
    }
    __syncthreads();                // cross-wave visibility of LDS K/V

    if (j0 + 32 < Lq) __builtin_prefetch(Hr + (size_t)(Lq - i0 + j0 + 17) * Dk, 0, 1);

    const __bf16* Kl = &ldsK[cur][0][0];   // [j][dk], ld=64
    const __bf16* Vl = &ldsV[cur][0][0];   // [dv][j], ld=32

    // ---- content scores (a)+(c): (hq+u)·hk^T, two 16-key tiles
    v8f S0 = {}, S1 = {};
    S0 = WMMA_BF16(aqu0, load_b_bf16(Kl, 64, lane), S0);
    S0 = WMMA_BF16(aqu1, load_b_bf16(Kl + 32, 64, lane), S0);
    S1 = WMMA_BF16(aqu0, load_b_bf16(Kl + 16 * 64, 64, lane), S1);
    S1 = WMMA_BF16(aqu1, load_b_bf16(Kl + 16 * 64 + 32, 64, lane), S1);
    // ---- position scores (b)+(d): unshifted panel over hr slab [base, base+48)
    const int base = Lq - i0 + j0 - 15;   // in [1, 2001]; +47 may touch pad row
#pragma unroll
    for (int tt = 0; tt < 3; ++tt) {
      v8f Pm = {};
      Pm = WMMA_BF16(aqv0, load_b_bf16(Hr + (size_t)(base + tt * 16) * Dk, Dk, lane), Pm);
      Pm = WMMA_BF16(aqv1, load_b_bf16(Hr + (size_t)(base + tt * 16) * Dk + 32, Dk, lane), Pm);
#pragma unroll
      for (int r = 0; r < 8; ++r) ldsP[wave][r + 8 * half][tt * 16 + n] = Pm[r];
    }
    // ---- Toeplitz gather + scale + online softmax (16-lane row groups)
    float p0v[8], p1v[8], alpha[8];
#pragma unroll
    for (int r = 0; r < 8; ++r) {
      const int m = r + 8 * half;
      float s0 = (S0[r] + ldsP[wave][m][15 + n - m]) * rcpT;
      float s1 = (S1[r] + ldsP[wave][m][31 + n - m]) * rcpT;
      float vmx = fmaxf(s0, s1);
#pragma unroll
      for (int d = 1; d < 16; d <<= 1) vmx = fmaxf(vmx, __shfl_xor(vmx, d, 32));
      const float mnew = fmaxf(mi[r], vmx);
      const float a0 = expf(s0 - mnew), a1 = expf(s1 - mnew);
      float rs = a0 + a1;
#pragma unroll
      for (int d = 1; d < 16; d <<= 1) rs += __shfl_xor(rs, d, 32);
      alpha[r] = expf(mi[r] - mnew);
      li[r] = li[r] * alpha[r] + rs;
      mi[r] = mnew;
      p0v[r] = a0; p1v[r] = a1;
    }
#pragma unroll
    for (int r = 0; r < 8; ++r) {
      const int m = r + 8 * half;
      ldsProb[wave][m][n]      = (__bf16)p0v[r];
      ldsProb[wave][m][16 + n] = (__bf16)p1v[r];
      O0[r] *= alpha[r]; O1[r] *= alpha[r]; O2[r] *= alpha[r]; O3[r] *= alpha[r];
    }
    // ---- PV: probs(16x32) @ hv(32x64), accumulate into rescaled O
    const v16bf ap = load_a_bf16(&ldsProb[wave][0][0], 32, lane);
    O0 = WMMA_BF16(ap, load_b_bf16(Vl + 0 * 16 * 32, 32, lane), O0);
    O1 = WMMA_BF16(ap, load_b_bf16(Vl + 1 * 16 * 32, 32, lane), O1);
    O2 = WMMA_BF16(ap, load_b_bf16(Vl + 2 * 16 * 32, 32, lane), O2);
    O3 = WMMA_BF16(ap, load_b_bf16(Vl + 3 * 16 * 32, 32, lane), O3);

    __syncthreads();                // protect LDS K/V before next overwrite
  }
  // ---- finalize and write concatenated heads [B*L, H*DV]
#pragma unroll
  for (int r = 0; r < 8; ++r) {
    const float inv = 1.0f / li[r];
    const size_t o = (size_t)(b * Lq + i0 + r + 8 * half) * Dm + h * Dk;
    heads[o + 0 * 16 + n] = (__bf16)(O0[r] * inv);
    heads[o + 1 * 16 + n] = (__bf16)(O1[r] * inv);
    heads[o + 2 * 16 + n] = (__bf16)(O2[r] * inv);
    heads[o + 3 * 16 + n] = (__bf16)(O3[r] * inv);
  }
}

// ---------------------------------------------------------------------------
// out = heads @ w_proj^T + q  (f32 residual buffer)
// ---------------------------------------------------------------------------
__global__ __launch_bounds__(128) void outproj_kernel(
    const __bf16* __restrict__ heads, const float* __restrict__ w_proj,
    const float* __restrict__ qres, float* __restrict__ resid) {
  const int lane = threadIdx.x & 31, wave = threadIdx.x >> 5;
  const int mt = blockIdx.x % ((Bq * Lq) / 16);
  const int nb = blockIdx.x / ((Bq * Lq) / 16);
  const int n0 = (nb * 4 + wave) * 16;
  const __bf16* A  = heads + (size_t)(mt * 16) * Dm;
  const float*  BT = w_proj + (size_t)n0 * Dm;   // w_proj[N=Dm][K=H*DV] == B^T
  v8f c = {};
  for (int k0 = 0; k0 < Dm; k0 += 32)
    c = WMMA_BF16(load_a_bf16(A + k0, Dm, lane), load_b_f32(BT + k0, Dm, lane), c);
  const int half = (lane >> 4) & 1, n = lane & 15;
#pragma unroll
  for (int r = 0; r < 8; ++r) {
    const size_t idx = (size_t)(mt * 16 + r + 8 * half) * Dm + n0 + n;
    resid[idx] = c[r] + qres[idx];
  }
}

// ---------------------------------------------------------------------------
// LayerNorm: unbiased std (/(D-1)), eps added to std (matches reference)
// ---------------------------------------------------------------------------
__global__ __launch_bounds__(256) void ln_kernel(
    const float* __restrict__ resid, const float* __restrict__ scale,
    const float* __restrict__ offset, float* __restrict__ out) {
  __shared__ float red[256];
  const int row = blockIdx.x;
  const float* x = resid + (size_t)row * Dm;
  float s = 0.f;
  for (int i = threadIdx.x; i < Dm; i += 256) s += x[i];
  red[threadIdx.x] = s; __syncthreads();
  for (int st = 128; st > 0; st >>= 1) {
    if (threadIdx.x < st) red[threadIdx.x] += red[threadIdx.x + st];
    __syncthreads();
  }
  const float mean = red[0] / (float)Dm; __syncthreads();
  float s2 = 0.f;
  for (int i = threadIdx.x; i < Dm; i += 256) { const float d = x[i] - mean; s2 += d * d; }
  red[threadIdx.x] = s2; __syncthreads();
  for (int st = 128; st > 0; st >>= 1) {
    if (threadIdx.x < st) red[threadIdx.x] += red[threadIdx.x + st];
    __syncthreads();
  }
  const float stdv = sqrtf(red[0] / (float)(Dm - 1)) + 1e-9f;
  for (int i = threadIdx.x; i < Dm; i += 256)
    out[(size_t)row * Dm + i] = scale[i] * (x[i] - mean) / stdv + offset[i];
}

// ---------------------------------------------------------------------------
extern "C" void kernel_launch(void* const* d_in, const int* in_sizes, int n_in,
                              void* d_out, int out_size, void* d_ws, size_t ws_size,
                              hipStream_t stream) {
  (void)in_sizes; (void)n_in; (void)out_size; (void)ws_size;
  const float* q        = (const float*)d_in[0];
  const float* k        = (const float*)d_in[1];
  const float* v        = (const float*)d_in[2];
  const float* Wq       = (const float*)d_in[3];
  const float* Wk       = (const float*)d_in[4];
  const float* Wv       = (const float*)d_in[5];
  const float* Wr       = (const float*)d_in[6];
  const float* u_w      = (const float*)d_in[7];
  const float* v_w      = (const float*)d_in[8];
  const float* w_proj   = (const float*)d_in[9];
  const float* ln_scale = (const float*)d_in[10];
  const float* ln_off   = (const float*)d_in[11];
  float* out = (float*)d_out;

  char* ws = (char*)d_ws;
  size_t off = 0;
  auto carve = [&](size_t bytes) -> char* {
    char* p = ws + off;
    off = (off + bytes + 255) & ~(size_t)255;
    return p;
  };
  float*  r_ws   = (float*) carve((size_t)Pp * Dm * 4);             // 8 MB
  __bf16* hq_u   = (__bf16*)carve((size_t)Bq * Hh * Lq * Dk * 2);   // 8 MB
  __bf16* hq_v   = (__bf16*)carve((size_t)Bq * Hh * Lq * Dk * 2);   // 8 MB
  __bf16* hk_ws  = (__bf16*)carve((size_t)Bq * Hh * Lq * Dk * 2);   // 8 MB
  __bf16* hvT    = (__bf16*)carve((size_t)Bq * Hh * Dk * Lq * 2);   // 8 MB
  __bf16* hr_ws  = (__bf16*)carve((size_t)Hh * HR_ROWS * Dk * 2);   // 4 MB
  __bf16* headsb = (__bf16*)carve((size_t)Bq * Lq * Dm * 2);        // 8 MB
  float*  resid  = (float*) carve((size_t)Bq * Lq * Dm * 4);        // 16 MB

  relpos_kernel<<<Pp, 256, 0, stream>>>(r_ws);
  proj_qkv_kernel<<<Bq * Hh * (Lq / 16), 128, 0, stream>>>(q, Wq, u_w, v_w, hq_u, hq_v, 0);
  proj_qkv_kernel<<<Bq * Hh * (Lq / 16), 128, 0, stream>>>(k, Wk, u_w, v_w, hk_ws, nullptr, 1);
  proj_qkv_kernel<<<Bq * Hh * (Lq / 16), 128, 0, stream>>>(v, Wv, u_w, v_w, hvT, nullptr, 2);
  proj_r_kernel<<<Hh * (Pp / 16), 128, 0, stream>>>(r_ws, Wr, hr_ws);
  flash_kernel<<<Bq * Hh * (Lq / 64), 128, 0, stream>>>(hq_u, hq_v, hk_ws, hvT, hr_ws, headsb);
  outproj_kernel<<<((Bq * Lq) / 16) * (Dm / 64), 128, 0, stream>>>(headsb, w_proj, q, resid);
  ln_kernel<<<Bq * Lq, 256, 0, stream>>>(resid, ln_scale, ln_off, out);
}
